// GCNLayer_42159398978043
// MI455X (gfx1250) — matrix operand order repaired and checked by reading, not compiled, not run on hardware
//
#include <hip/hip_runtime.h>

typedef __attribute__((ext_vector_type(16))) _Float16 v16h;
typedef __attribute__((ext_vector_type(8)))  _Float16 v8h;
typedef __attribute__((ext_vector_type(4)))  _Float16 v4h;
typedef __attribute__((ext_vector_type(8)))  float    v8f;
typedef __attribute__((ext_vector_type(4)))  float    v4f;

#define NROWS 8192
#define INF   256
#define OUTF  256

static __device__ __forceinline__ v16h cat8(v8h lo, v8h hi) {
  return __builtin_shufflevector(lo, hi, 0,1,2,3,4,5,6,7,8,9,10,11,12,13,14,15);
}

// ---------------- Kernel 1: deg[i] = 1 + sum_j A[i][j]; ds[i] = rsqrt(deg) ----
__global__ __launch_bounds__(256) void deg_kernel(const float* __restrict__ A,
                                                  float* __restrict__ ds) {
  const int wave = threadIdx.x >> 5;
  const int lane = threadIdx.x & 31;
  const int row  = blockIdx.x * 8 + wave;
  const v4f* arow = (const v4f*)(A + (size_t)row * NROWS);
  float s = 0.f;
  for (int i = lane; i < NROWS / 4; i += 32) {
    v4f v = __builtin_nontemporal_load(arow + i);   // NT: A is stream-once
    s += (v.x + v.y) + (v.z + v.w);
  }
  #pragma unroll
  for (int off = 16; off > 0; off >>= 1) s += __shfl_xor(s, off, 32);
  if (lane == 0) ds[row] = rsqrtf(s + 1.0f);
}

// ------- Kernel 2: XsT[n][k] = f16(ds[k] * X[k][n])  (scale + transpose) -----
__global__ __launch_bounds__(256) void xt_kernel(const float* __restrict__ X,
                                                 const float* __restrict__ ds,
                                                 _Float16* __restrict__ XsT) {
  __shared__ _Float16 tile[32][33];
  const int k0 = blockIdx.x * 32;
  const int n0 = blockIdx.y * 32;
  const int tx = threadIdx.x;  // 0..31
  const int ty = threadIdx.y;  // 0..7
  #pragma unroll
  for (int j = 0; j < 32; j += 8) {
    const int k = k0 + ty + j;
    tile[ty + j][tx] = (_Float16)(ds[k] * X[(size_t)k * INF + n0 + tx]);
  }
  __syncthreads();
  #pragma unroll
  for (int j = 0; j < 32; j += 8) {
    const int n = n0 + ty + j;
    XsT[(size_t)n * NROWS + k0 + tx] = tile[tx][ty + j];
  }
}

// ---------------- Kernel 3: Wh = f16(W) --------------------------------------
__global__ __launch_bounds__(256) void wh_kernel(const float* __restrict__ W,
                                                 _Float16* __restrict__ Wh) {
  const int i = blockIdx.x * 256 + threadIdx.x;
  Wh[i] = (_Float16)W[i];
}

// ---------------- Kernel 4: h = diag(ds) * (A + I) * XsT^T  (f16 WMMA) -------
// One block = 16-row strip x all 256 cols. 8 waves, 2 N-tiles per wave.
// 64 K-values staged in LDS per barrier (double buffered): 4 WMMAs per sync.
__global__ __launch_bounds__(256) void gcn_agg_kernel(
    const float* __restrict__ A, const _Float16* __restrict__ XsT,
    const float* __restrict__ ds, _Float16* __restrict__ hs) {
  __shared__ __align__(16) _Float16 atile[2][16 * 64];

  const int tid    = threadIdx.x;
  const int wave   = tid >> 5;
  const int lane   = tid & 31;
  const int lanelo = lane & 15;
  const int m0     = blockIdx.x * 16;

  // cooperative A-tile staging coords: 4 consecutive floats per thread
  const int srow = tid >> 4;         // 0..15
  const int scol = (tid & 15) * 4;   // 0..60
  const float* aptr = A + (size_t)(m0 + srow) * NROWS + scol;

  // A-fragment LDS column offsets within a 32-K subtile (ISA 7.12.2)
  const int koffA0 = (lane < 16) ? 0 : 8;
  const int koffA1 = koffA0 + 16;

  // B-fragment pointers: lane n<16 -> col n, K 0..7,16..23; lane>=16 -> K 8..15,24..31
  const int koffB = (lane < 16) ? 0 : 8;
  const _Float16* bp0 = XsT + (size_t)(wave * 32 + lanelo) * NROWS + koffB;
  const _Float16* bp1 = XsT + (size_t)(wave * 32 + 16 + lanelo) * NROWS + koffB;

  v8f c0 = {};
  v8f c1 = {};

  for (int kt = 0; kt < NROWS / 64; ++kt) {
    const int k0  = kt * 64;
    const int buf = kt & 1;

    // stage 16x64 A tile: NT f32 load -> (+I on diagonal) -> f16 -> LDS b64
    v4f av = __builtin_nontemporal_load((const v4f*)(aptr + k0));
    if (kt + 1 < NROWS / 64) __builtin_prefetch(aptr + k0 + 64, 0, 0);
    v4h hv;
    #pragma unroll
    for (int e = 0; e < 4; ++e) {
      float v = av[e] + (((m0 + srow) == (k0 + scol + e)) ? 1.0f : 0.0f);
      hv[e] = (_Float16)v;
    }
    *(v4h*)&atile[buf][srow * 64 + scol] = hv;
    __syncthreads();

    #pragma unroll
    for (int s = 0; s < 2; ++s) {
      const int kk = k0 + s * 32;
      // A fragment from LDS
      v8h alo = *(const v8h*)&atile[buf][lanelo * 64 + s * 32 + koffA0];
      v8h ahi = *(const v8h*)&atile[buf][lanelo * 64 + s * 32 + koffA1];
      v16h af = cat8(alo, ahi);
      // B fragments from global (L2-resident XsT; default temporal hint)
      v16h b0 = cat8(*(const v8h*)(bp0 + kk), *(const v8h*)(bp0 + kk + 16));
      v16h b1 = cat8(*(const v8h*)(bp1 + kk), *(const v8h*)(bp1 + kk + 16));
      c0 = __builtin_amdgcn_wmma_f32_16x16x32_f16(false, af, false, b0,
                                                  (short)0, c0, false, false);
      c1 = __builtin_amdgcn_wmma_f32_16x16x32_f16(false, af, false, b1,
                                                  (short)0, c1, false, false);
    }
  }

  // epilogue: scale rows by ds[i], store h as f16
  const int roff = (lane < 16) ? 0 : 8;
  float4 d0 = *(const float4*)(ds + m0 + roff);
  float4 d1 = *(const float4*)(ds + m0 + roff + 4);
  float dsr[8] = {d0.x, d0.y, d0.z, d0.w, d1.x, d1.y, d1.z, d1.w};
  _Float16* hrow = hs + (size_t)(m0 + roff) * INF + lanelo;
  #pragma unroll
  for (int r = 0; r < 8; ++r) {
    hrow[(size_t)r * INF + wave * 32]      = (_Float16)(c0[r] * dsr[r]);
    hrow[(size_t)r * INF + wave * 32 + 16] = (_Float16)(c1[r] * dsr[r]);
  }
}

// ---------------- Kernel 5: out = h @ W^T + b  (f16 WMMA) --------------------
// Row-major W[o][k] is exactly the contiguous layout B-fragment column o needs.
__global__ __launch_bounds__(256) void gcn_lin_kernel(
    const _Float16* __restrict__ hs, const _Float16* __restrict__ Wh,
    const float* __restrict__ bias, float* __restrict__ out) {
  const int tid    = threadIdx.x;
  const int wave   = tid >> 5;
  const int lane   = tid & 31;
  const int lanelo = lane & 15;
  const int m0     = blockIdx.x * 16;
  const int n0     = wave * 32;

  const int koffA = (lane < 16) ? 0 : 8;
  const _Float16* ap  = hs + (size_t)(m0 + lanelo) * INF + koffA;
  const int koffB = (lane < 16) ? 0 : 8;
  const _Float16* bp0 = Wh + (size_t)(n0 + lanelo) * INF + koffB;
  const _Float16* bp1 = Wh + (size_t)(n0 + 16 + lanelo) * INF + koffB;

  v8f c0 = {};
  v8f c1 = {};
  #pragma unroll
  for (int k0 = 0; k0 < INF; k0 += 32) {
    v16h af = cat8(*(const v8h*)(ap + k0), *(const v8h*)(ap + k0 + 16));
    v16h b0 = cat8(*(const v8h*)(bp0 + k0), *(const v8h*)(bp0 + k0 + 16));
    v16h b1 = cat8(*(const v8h*)(bp1 + k0), *(const v8h*)(bp1 + k0 + 16));
    c0 = __builtin_amdgcn_wmma_f32_16x16x32_f16(false, af, false, b0,
                                                (short)0, c0, false, false);
    c1 = __builtin_amdgcn_wmma_f32_16x16x32_f16(false, af, false, b1,
                                                (short)0, c1, false, false);
  }

  const float bias0 = bias[n0 + lanelo];
  const float bias1 = bias[n0 + 16 + lanelo];
  const int roff = (lane < 16) ? 0 : 8;
  float* orow = out + (size_t)(m0 + roff) * OUTF + lanelo;
  #pragma unroll
  for (int r = 0; r < 8; ++r) {
    orow[(size_t)r * OUTF + n0]      = c0[r] + bias0;
    orow[(size_t)r * OUTF + n0 + 16] = c1[r] + bias1;
  }
}

extern "C" void kernel_launch(void* const* d_in, const int* in_sizes, int n_in,
                              void* d_out, int out_size, void* d_ws, size_t ws_size,
                              hipStream_t stream) {
  (void)in_sizes; (void)n_in; (void)out_size; (void)ws_size;
  const float* X = (const float*)d_in[0];
  const float* A = (const float*)d_in[1];
  const float* W = (const float*)d_in[2];
  const float* b = (const float*)d_in[3];

  char* ws = (char*)d_ws;
  float*    dsv = (float*)ws;                                   // 32 KB
  _Float16* XsT = (_Float16*)(ws + 32768);                      // 4 MB
  _Float16* Wh  = (_Float16*)(ws + 32768 + 4194304);            // 128 KB
  _Float16* hs  = (_Float16*)(ws + 32768 + 4194304 + 131072);   // 4 MB

  deg_kernel<<<NROWS / 8, 256, 0, stream>>>(A, dsv);
  xt_kernel<<<dim3(NROWS / 32, INF / 32), dim3(32, 8), 0, stream>>>(X, dsv, XsT);
  wh_kernel<<<(OUTF * INF) / 256, 256, 0, stream>>>(W, Wh);
  gcn_agg_kernel<<<NROWS / 16, 256, 0, stream>>>(A, XsT, dsv, hs);
  gcn_lin_kernel<<<NROWS / 16, 256, 0, stream>>>(hs, Wh, b, (float*)d_out);
}
